// SpMiddleNoDownsampleXY_4750233829983
// MI455X (gfx1250) — matrix operand order repaired and checked by reading, not compiled, not run on hardware
//
#include <hip/hip_runtime.h>
#include <stdint.h>

typedef __attribute__((ext_vector_type(16))) _Float16 v16h;
typedef __attribute__((ext_vector_type(8)))  _Float16 v8h;
typedef __attribute__((ext_vector_type(8)))  float    v8f;

#define HDIM 256
#define WDIM 256
#define BN_EPS 1e-3f

// ---------------------------------------------------------------------------
// Layer parameter block (passed by value as kernarg)
// ---------------------------------------------------------------------------
struct ConvP {
  const _Float16* __restrict__ in;   // [Din][H][W][Cin] f16 channels-last
  _Float16* __restrict__ out;        // [Dout][H][W][Cout]
  const _Float16* __restrict__ wgt;  // preconverted [tap][kc][n][32] f16
  const float* gamma;
  const float* beta;
  const float* mean;
  const float* var;
  const _Float16* mIn;               // [Din][H][W] f16 mask (0/1)
  _Float16* mOut;                    // [Dout][H][W] (sp layers only)
  int Din, Dout, Cin, Cout;
  int kd, kh, kw, sz, pd, ph, pw;
  int subm;                          // 1 = submanifold (reuse mask), 0 = sp (dilate)
};

// ---------------------------------------------------------------------------
// Zero-fill (graph-capture-safe replacement for hipMemset)
// ---------------------------------------------------------------------------
__global__ void fill_zero_u32(uint32_t* p, size_t n) {
  size_t i = (size_t)blockIdx.x * blockDim.x + threadIdx.x;
  size_t stride = (size_t)gridDim.x * blockDim.x;
  for (; i < n; i += stride) p[i] = 0u;
}

// ---------------------------------------------------------------------------
// Scatter 40000 x 16 f32 voxel features into dense f16 volume + mask
// ---------------------------------------------------------------------------
__global__ void scatter16(const float* __restrict__ feat,
                          const int* __restrict__ coors,
                          _Float16* __restrict__ dense,
                          _Float16* __restrict__ mask, int nvox) {
  int i = blockIdx.x * blockDim.x + threadIdx.x;
  if (i >= nvox) return;
  int z = coors[i * 4 + 1];
  int y = coors[i * 4 + 2];
  int x = coors[i * 4 + 3];
  size_t base = ((size_t)z * HDIM + y) * WDIM + x;
  _Float16* dst = dense + base * 16;
#pragma unroll
  for (int c = 0; c < 16; ++c) dst[c] = (_Float16)feat[i * 16 + c];
  mask[base] = (_Float16)1.0f;
}

// ---------------------------------------------------------------------------
// One-time per layer: f32 [Cout][Cin][kd][kh][kw] -> f16 [tap][kc][n][32]
// (K zero-padded to 32-channel chunks; B-tile per (tap,kc) is contiguous)
// ---------------------------------------------------------------------------
__global__ void prep_weights(const float* __restrict__ wgt, _Float16* __restrict__ out,
                             int Cin, int Cout, int kd, int kh, int kw) {
  int kcn = (Cin + 31) >> 5;
  int khw = kh * kw;
  int total = kd * khw * kcn * Cout * 32;
  int idx = blockIdx.x * blockDim.x + threadIdx.x;
  if (idx >= total) return;
  int k = idx & 31;
  int n = (idx >> 5) % Cout;
  int rest = idx / (32 * Cout);
  int kc = rest % kcn;
  int tap = rest / kcn;
  int dz = tap / khw;
  int r = tap % khw;
  int dy = r / kw;
  int dx = r % kw;
  int ci = kc * 32 + k;
  float v = 0.0f;
  if (ci < Cin)
    v = wgt[(((size_t)n * Cin + ci) * kd + dz) * khw + dy * kw + dx];
  out[idx] = (_Float16)v;
}

// ---------------------------------------------------------------------------
// Fused layer: implicit-GEMM conv3d (WMMA f32 = f16 x f16) + BN + ReLU + mask.
// Block = 128 threads (4 wave32), each wave owns one 16-channel N-tile.
// Tile = 16 consecutive output voxels along W x all Cout channels.
// Double-buffered LDS pipeline over the flattened (tap,kc) iteration space:
// iteration `it` stages it+1's A/B while issuing it's WMMA -> one barrier/iter.
// Whole-tile skip when the 16-voxel sparsity mask is empty.
// ---------------------------------------------------------------------------
__global__ __launch_bounds__(128) void conv3d_wmma_bn_relu(ConvP p) {
  const int x0 = blockIdx.x << 4;
  const int yo = blockIdx.y;
  const int zo = blockIdx.z;
  const int tid = threadIdx.x;
  const int lane = tid & 31;
  const int wv = tid >> 5;
  const int hi = lane >> 4;     // lane half (0: lanes 0-15, 1: lanes 16-31)
  const int row = lane & 15;    // M index for A fragment

  // A stride padded to 80B so per-lane fragment reads are aligned ds_load_b128
  __shared__ __align__(16) _Float16 sA[2][16][40];   // 16 voxels x 32 K
  __shared__ __align__(16) _Float16 sB[2][64 * 32];  // [n][k] contiguous
  __shared__ __align__(16) _Float16 sOut[16 * 64];
  __shared__ float sMask[16];
  __shared__ int sAny;

  if (tid == 0) sAny = 0;
  __syncthreads();

  // ---- sparsity mask for the 16 output voxels of this tile ----
  if (tid < 16) {
    float mv = 0.0f;
    if (p.subm) {
      mv = (float)p.mIn[((size_t)zo * HDIM + yo) * WDIM + x0 + tid];
    } else {
      for (int dz = 0; dz < p.kd; ++dz) {
        int zi = zo * p.sz + dz - p.pd;
        if (zi < 0 || zi >= p.Din) continue;
        for (int dy = 0; dy < p.kh; ++dy) {
          int yi = yo + dy - p.ph;
          if (yi < 0 || yi >= HDIM) continue;
          for (int dx = 0; dx < p.kw; ++dx) {
            int xi = x0 + tid + dx - p.pw;
            if (xi < 0 || xi >= WDIM) continue;
            if ((float)p.mIn[((size_t)zi * HDIM + yi) * WDIM + xi] > 0.0f)
              mv = 1.0f;
          }
        }
      }
      p.mOut[((size_t)zo * HDIM + yo) * WDIM + x0 + tid] = (_Float16)mv;
    }
    sMask[tid] = mv;
    if (mv > 0.0f) atomicOr(&sAny, 1);
  }
  __syncthreads();

  const size_t obase = (((size_t)zo * HDIM + yo) * WDIM + x0) * p.Cout;
  const int totalOut = 16 * p.Cout;

  // ---- sparse tile skip: all 16 outputs masked off -> store zeros ----
  if (!sAny) {
    int idx = tid * 8;
    if (idx < totalOut) {
      uint4 z4 = {0u, 0u, 0u, 0u};
      *(uint4*)(p.out + obase + idx) = z4;
    }
    return;
  }

  const int ntile = wv;                       // N-tile per wave
  const bool act = (ntile * 16) < p.Cout;     // wave-uniform
  const int col = ntile * 16 + (lane & 15);   // output channel of this lane

  float scale = 0.0f, shift = 0.0f;
  if (act) {
    scale = p.gamma[col] * rsqrtf(p.var[col] + BN_EPS);
    shift = p.beta[col] - p.mean[col] * scale;
  }

  const int kcn = (p.Cin + 31) >> 5;          // K chunks of 32 input channels
  const int khw = p.kh * p.kw;
  const int T = p.kd * khw * kcn;             // flattened (tap,kc) iterations
  const int btile = 32 * p.Cout;              // halves per contiguous B tile

  // ---- stage A (16x32 activations) + B (32xCout weights) for step `it` ----
  auto stage = [&](int it, int buf) {
    int tap = it / kcn;
    int kc = it - tap * kcn;
    int dz = tap / khw;
    int r = tap - dz * khw;
    int dy = r / p.kw;
    int dx = r - dy * p.kw;
    int zi = zo * p.sz + dz - p.pd;
    int yi = yo + dy - p.ph;
    bool yok = (zi >= 0) && (zi < p.Din) && (yi >= 0) && (yi < HDIM);
    // A: each thread loads 4 contiguous halves of one voxel's channel chunk
    int v = tid >> 3;
    int cg = (tid & 7) * 4;
    int c = kc * 32 + cg;
    int xi = x0 + v + dx - p.pw;
    unsigned long long val = 0ull;
    if (yok && xi >= 0 && xi < WDIM && c < p.Cin)
      val = *(const unsigned long long*)(
          p.in + (((size_t)zi * HDIM + yi) * WDIM + xi) * p.Cin + c);
    *(unsigned long long*)&sA[buf][v][cg] = val;
    // B: contiguous preconverted tile -> vectorized b128 copies
    const _Float16* src = p.wgt + (size_t)it * btile;
    for (int idx = tid * 8; idx < btile; idx += 1024)
      *(uint4*)&sB[buf][idx] = *(const uint4*)(src + idx);
    // keep the weight stream warm in L2 (global_prefetch_b8)
    if (tid == 0 && it + 2 < T)
      __builtin_prefetch(p.wgt + (size_t)(it + 2) * btile, 0, 0);
  };

  v8f acc = {};
  stage(0, 0);
  __syncthreads();
  for (int it = 0; it < T; ++it) {
    int cur = it & 1;
    if (it + 1 < T) stage(it + 1, cur ^ 1);
    if (act) {
      // A fragment: lane half hi holds K = hi*8..hi*8+7 and +16 (ISA layout)
      v8h a0 = *(const v8h*)&sA[cur][row][hi * 8];
      v8h a1 = *(const v8h*)&sA[cur][row][16 + hi * 8];
      // B fragment: lane half hi holds K = hi*8..hi*8+7 and +16 of column `col`
      v8h b0 = *(const v8h*)&sB[cur][col * 32 + hi * 8];
      v8h b1 = *(const v8h*)&sB[cur][col * 32 + 16 + hi * 8];
      v16h a, b;
#pragma unroll
      for (int j = 0; j < 8; ++j) {
        a[j] = a0[j];
        a[j + 8] = a1[j];
        b[j] = b0[j];
        b[j + 8] = b1[j];
      }
      acc = __builtin_amdgcn_wmma_f32_16x16x32_f16(false, a, false, b, (short)0,
                                                   acc, false, false);
    }
    __syncthreads();
  }

  // ---- fused BN + ReLU + mask, staged through LDS for b128 stores ----
  if (act) {
#pragma unroll
    for (int r = 0; r < 8; ++r) {
      int m = r + hi * 8;                      // voxel index (D-matrix layout)
      float v = acc[r] * scale + shift;
      v = v > 0.0f ? v : 0.0f;
      v *= sMask[m];
      sOut[m * p.Cout + col] = (_Float16)v;
    }
  }
  __syncthreads();
  int idx = tid * 8;
  if (idx < totalOut)
    *(uint4*)(p.out + obase + idx) = *(const uint4*)(&sOut[idx]);
}

// ---------------------------------------------------------------------------
// Final layout conversion: f16 [Df][H][W][64] -> f32 [(c*Df+d)][H][W]
// ---------------------------------------------------------------------------
__global__ void finalize_out(const _Float16* __restrict__ in, float* __restrict__ out,
                             int Df, size_t total) {
  size_t i = (size_t)blockIdx.x * blockDim.x + threadIdx.x;
  if (i >= total) return;
  int x = (int)(i & 255);
  int y = (int)((i >> 8) & 255);
  size_t r = i >> 16;
  int d = (int)(r % (size_t)Df);
  int c = (int)(r / (size_t)Df);
  out[i] = (float)in[(((size_t)d * HDIM + y) * WDIM + x) * 64 + c];
}

// ---------------------------------------------------------------------------
// Host driver
// ---------------------------------------------------------------------------
extern "C" void kernel_launch(void* const* d_in, const int* in_sizes, int n_in,
                              void* d_out, int out_size, void* d_ws, size_t ws_size,
                              hipStream_t stream) {
  (void)in_sizes; (void)n_in; (void)out_size; (void)ws_size;
  const float* vfeat = (const float*)d_in[0];
  const int* coors = (const int*)d_in[1];
  // d_in[2] = batch_size (always 1)
  // d_in[3..13]=weights, [14..24]=gammas, [25..35]=betas, [36..46]=means, [47..57]=vars

  // Workspace: 2 ping-pong f16 feature buffers + 2 f16 masks + converted weights
  const size_t featBytes = (size_t)9 * HDIM * WDIM * 64 * 2;  // >= 17*256*256*32 f16
  const size_t maskBytes = (size_t)17 * HDIM * WDIM * 2;
  const size_t wslotHalves = (size_t)27 * 2 * 64 * 32;        // max taps*kcn*Cout*32
  char* ws = (char*)d_ws;
  _Float16* feat[2] = {(_Float16*)ws, (_Float16*)(ws + featBytes)};
  _Float16* maskb[2] = {(_Float16*)(ws + 2 * featBytes),
                        (_Float16*)(ws + 2 * featBytes + maskBytes)};
  _Float16* wcvtBase = (_Float16*)(ws + 2 * featBytes + 2 * maskBytes);

  struct L { int typ, kd, kh, kw, sz, pd, ph, pw, cin, cout; };
  const L Ls[11] = {
      {0, 3, 3, 3, 1, 1, 1, 1, 16, 32}, {0, 3, 3, 3, 1, 1, 1, 1, 32, 32},
      {1, 3, 3, 3, 2, 1, 1, 1, 32, 64}, {0, 3, 3, 3, 1, 1, 1, 1, 64, 64},
      {0, 3, 3, 3, 1, 1, 1, 1, 64, 64}, {0, 3, 3, 3, 1, 1, 1, 1, 64, 64},
      {1, 3, 3, 3, 2, 0, 1, 1, 64, 64}, {0, 3, 3, 3, 1, 1, 1, 1, 64, 64},
      {0, 3, 3, 3, 1, 1, 1, 1, 64, 64}, {0, 3, 3, 3, 1, 1, 1, 1, 64, 64},
      {1, 3, 1, 1, 2, 0, 0, 0, 64, 64}};

  // One-time weight conversion (independent of the feature pipeline)
  for (int i = 0; i < 11; ++i) {
    const L& l = Ls[i];
    int kcn = (l.cin + 31) >> 5;
    int total = l.kd * l.kh * l.kw * kcn * l.cout * 32;
    prep_weights<<<(total + 255) / 256, 256, 0, stream>>>(
        (const float*)d_in[3 + i], wcvtBase + (size_t)i * wslotHalves,
        l.cin, l.cout, l.kd, l.kh, l.kw);
  }

  // Zero layer-0 feature region (D17 x C16) and initial mask, then scatter
  fill_zero_u32<<<4096, 256, 0, stream>>>((uint32_t*)feat[0],
                                          ((size_t)17 * HDIM * WDIM * 16) / 2);
  fill_zero_u32<<<2048, 256, 0, stream>>>((uint32_t*)maskb[0],
                                          ((size_t)17 * HDIM * WDIM) / 2);
  scatter16<<<(40000 + 255) / 256, 256, 0, stream>>>(vfeat, coors, feat[0],
                                                     maskb[0], 40000);

  int din = 17, cb = 0, mb = 0;
  for (int i = 0; i < 11; ++i) {
    const L& l = Ls[i];
    int dout = (din + 2 * l.pd - l.kd) / l.sz + 1;
    ConvP p;
    p.in = feat[cb];
    p.out = feat[cb ^ 1];
    p.wgt = wcvtBase + (size_t)i * wslotHalves;
    p.gamma = (const float*)d_in[14 + i];
    p.beta = (const float*)d_in[25 + i];
    p.mean = (const float*)d_in[36 + i];
    p.var = (const float*)d_in[47 + i];
    p.mIn = maskb[mb];
    p.mOut = maskb[mb ^ 1];  // written only when typ==sp
    p.Din = din; p.Dout = dout; p.Cin = l.cin; p.Cout = l.cout;
    p.kd = l.kd; p.kh = l.kh; p.kw = l.kw; p.sz = l.sz;
    p.pd = l.pd; p.ph = l.ph; p.pw = l.pw;
    p.subm = (l.typ == 0);
    dim3 grid(WDIM / 16, HDIM, dout);
    conv3d_wmma_bn_relu<<<grid, 128, 0, stream>>>(p);
    if (l.typ) mb ^= 1;
    cb ^= 1;
    din = dout;
  }

  size_t total = (size_t)64 * din * HDIM * WDIM;
  finalize_out<<<(int)((total + 255) / 256), 256, 0, stream>>>(
      feat[cb], (float*)d_out, din, total);
}
